// EncoderLayer_86011015069993
// MI455X (gfx1250) — compile-verified, matrix operand order checked by reading, not run on hardware
//
#include <hip/hip_runtime.h>
#include <hip/hip_bf16.h>

// ---------------------------------------------------------------------------
// Problem constants (fixed by the reference)
// ---------------------------------------------------------------------------
#define S_   2048
#define B_   2
#define E_   1024
#define H_   16
#define HD_  64
#define FFN_ 4096
#define NTOK (S_ * B_)          // 4096 token rows
#define QBLK 64                 // q rows per flash block
#define NQB  (S_ / QBLK)        // 32

typedef __attribute__((ext_vector_type(16))) __bf16 v16bf;
typedef __attribute__((ext_vector_type(8)))  float  v8f;

union Frag16 {                   // 16 bf16 = 8 dwords (A/B operand of wmma)
    v16bf b;
    unsigned short u[16];
    unsigned int   d[8];
};
union U4 {
    uint4 v;
    unsigned short s[8];
    unsigned int   u[4];
};

__device__ __forceinline__ unsigned short f32_to_bf16(float f) {
    unsigned u = __builtin_bit_cast(unsigned, f);
    u = (u + 0x7FFFu + ((u >> 16) & 1u)) >> 16;     // RNE
    return (unsigned short)u;
}

// ---------------------------------------------------------------------------
// Tensor Data Mover: 2-D tile load (global -> LDS), descriptor per ISA 8.3/8.4
// ---------------------------------------------------------------------------
#if __has_builtin(__builtin_amdgcn_tensor_load_to_lds) && \
    __has_builtin(__builtin_amdgcn_s_wait_tensorcnt)
#define USE_TDM 1
#else
#define USE_TDM 0
#endif

#if USE_TDM
typedef unsigned int v4u_t __attribute__((ext_vector_type(4)));
typedef int          v8i_t __attribute__((ext_vector_type(8)));
typedef int          v4i_t __attribute__((ext_vector_type(4)));

// tileX elements contiguous (dim0), tileY rows, row stride = strideElems,
// 2-byte elements (bf16).  All arguments must be workgroup-uniform.
__device__ __forceinline__ void tdm_load_tile_2d(void* ldsPtr, const void* gPtr,
                                                 unsigned tileX, unsigned tileY,
                                                 unsigned tdim0, unsigned tdim1,
                                                 unsigned strideElems) {
    unsigned long long ga = (unsigned long long)gPtr;
    unsigned ldsOff = (unsigned)(unsigned long long)ldsPtr;  // LDS aperture low32
    v4u_t g0;
    g0[0] = 1u;                                   // count=1, user descriptor
    g0[1] = ldsOff;                               // lds_addr
    g0[2] = (unsigned)ga;                         // global_addr[31:0]
    g0[3] = (unsigned)((ga >> 32) & 0x1FFFFFFu)   // global_addr[56:32]
          | (2u << 30);                           // type = 2 ("image")
    v8i_t g1;
    g1[0] = (int)(1u << 16);                      // data_size = 1 -> 2 bytes
    g1[1] = (int)((tdim0 & 0xFFFFu) << 16);       // tensor_dim0[15:0]
    g1[2] = (int)((tdim0 >> 16) | ((tdim1 & 0xFFFFu) << 16));
    g1[3] = (int)((tdim1 >> 16) | (tileX << 16)); // tile_dim0
    g1[4] = (int)(tileY & 0xFFFFu);               // tile_dim1 (tile_dim2 = 0)
    g1[5] = (int)strideElems;                     // tensor_dim0_stride[31:0]
    g1[6] = 0;                                    // stride hi + dim1_stride lo
    g1[7] = 0;
    v4i_t z = {0, 0, 0, 0};
#if __clang_major__ >= 23
    v8i_t z8 = {0, 0, 0, 0, 0, 0, 0, 0};
    __builtin_amdgcn_tensor_load_to_lds(g0, g1, z, z, z8, 0);
#else
    __builtin_amdgcn_tensor_load_to_lds(g0, g1, z, z, 0);
#endif
}
#endif

// ---------------------------------------------------------------------------
// fp32 -> bf16 elementwise convert (grid-stride)
// ---------------------------------------------------------------------------
__global__ void cvt_f32_bf16(const float* __restrict__ in,
                             unsigned short* __restrict__ out, int n) {
    for (int i = blockIdx.x * blockDim.x + threadIdx.x; i < n;
         i += gridDim.x * blockDim.x)
        out[i] = f32_to_bf16(in[i]);
}

// ---------------------------------------------------------------------------
// Tiled bf16 WMMA GEMM:  C(M x N) = A(M x K) * W(K x N) + bias
//   block tile 128x128, BK=32, 256 threads = 8 wave32 waves (4x2 wave grid),
//   each wave owns 32x64 (2x4 accumulators -> 8 WMMAs per K-step).
//   Double-buffered LDS; A tile staged by TDM (tensor_load_to_lds), B tile
//   staged transposed by threads with paired-row b32 stores.
// mode 0: C fp32 row-major            (out = float*)
// mode 1: bf16 scatter to (B,H,S,HD)  (out = ushort*), value=(v+bias)*scale
// mode 2: relu -> bf16 row-major      (out = ushort*)
// ---------------------------------------------------------------------------
__global__ __launch_bounds__(256)
void gemm_bf16(const unsigned short* __restrict__ A,
               const unsigned short* __restrict__ W,
               const float* __restrict__ bias,
               int M, int N, int K, int mode, float scale,
               void* __restrict__ out) {
    __shared__ __align__(16) unsigned short ldsA[2][128 * 32];   // [m][k]
    __shared__ __align__(16) unsigned short ldsBT[2][128 * 32];  // [n][k]

    const int tid  = threadIdx.x;
    const int w    = tid >> 5;
    const int lane = tid & 31;
    const int half = lane >> 4;
    const int l16  = lane & 15;
    const int wm   = w >> 1;          // 0..3 -> row offset 32*wm
    const int wn   = w & 1;           // 0..1 -> col offset 64*wn

    const int n0 = blockIdx.x * 128;
    const int m0 = blockIdx.y * 128;

    // B staging assignment: 16 k-row pairs x 16 col-chunks (8 cols each)
    const int bk0 = (tid >> 4) * 2;   // k row pair base: 0,2,..,30
    const int bc  = tid & 15;         // col chunk: 8 columns each

    v8f acc[2][4];
#pragma unroll
    for (int ms = 0; ms < 2; ++ms)
#pragma unroll
        for (int ns = 0; ns < 4; ++ns) acc[ms][ns] = (v8f){};

    const int ksteps = K >> 5;

    // ---- prologue: stage k-step 0 into buffer 0
    {
        U4 b0, b1;
        b0.v = *(const uint4*)&W[(size_t)(bk0 + 0) * N + n0 + bc * 8];
        b1.v = *(const uint4*)&W[(size_t)(bk0 + 1) * N + n0 + bc * 8];
#pragma unroll
        for (int e = 0; e < 8; ++e)
            *(unsigned int*)&ldsBT[0][(bc * 8 + e) * 32 + bk0] =
                (unsigned)b0.s[e] | ((unsigned)b1.s[e] << 16);
#if USE_TDM
        if (tid < 32)
            tdm_load_tile_2d(&ldsA[0][0], &A[(size_t)m0 * K], 32, 128,
                             (unsigned)K, (unsigned)M, (unsigned)K);
#else
#pragma unroll
        for (int it = 0; it < 2; ++it) {
            int chunk = tid + it * 256;          // 0..511
            int row = chunk >> 2, cc = chunk & 3;
            U4 va;
            va.v = *(const uint4*)&A[(size_t)(m0 + row) * K + cc * 8];
            *(uint4*)&ldsA[0][row * 32 + cc * 8] = va.v;
        }
#endif
    }

    for (int kt = 0; kt < ksteps; ++kt) {
        const int cur = kt & 1, nxt = cur ^ 1;
        const bool doStage = (kt + 1) < ksteps;

#if USE_TDM
        if (tid < 32) __builtin_amdgcn_s_wait_tensorcnt(0);
#endif
        __syncthreads();                          // buf[cur] ready

        // ---- issue next-tile staging (loads into regs / TDM) --------------
        U4 b0, b1;
#if !USE_TDM
        U4 aR[2];
#endif
        if (doStage) {
            const int ktn = kt + 1;
            b0.v = *(const uint4*)&W[(size_t)(ktn * 32 + bk0 + 0) * N + n0 + bc * 8];
            b1.v = *(const uint4*)&W[(size_t)(ktn * 32 + bk0 + 1) * N + n0 + bc * 8];
#if USE_TDM
            if (tid < 32)
                tdm_load_tile_2d(&ldsA[nxt][0], &A[(size_t)m0 * K + ktn * 32],
                                 32, 128, (unsigned)K, (unsigned)M, (unsigned)K);
#else
#pragma unroll
            for (int it = 0; it < 2; ++it) {
                int chunk = tid + it * 256;
                int row = chunk >> 2, cc = chunk & 3;
                aR[it].v = *(const uint4*)&A[(size_t)(m0 + row) * K + ktn * 32 + cc * 8];
            }
#endif
        }

        // ---- compute: 2 A-frags x 4 B-frags = 8 WMMAs ---------------------
        Frag16 fa[2];
#pragma unroll
        for (int ms = 0; ms < 2; ++ms) {
            int m = 32 * wm + 16 * ms + l16;
#pragma unroll
            for (int p = 0; p < 8; ++p) {
                int kk = (p >> 2) * 16 + half * 8 + (p & 3) * 2;
                fa[ms].d[p] = *(const unsigned int*)&ldsA[cur][m * 32 + kk];
            }
        }
#pragma unroll
        for (int ns = 0; ns < 4; ++ns) {
            Frag16 fb;
            int n = 64 * wn + 16 * ns + l16;
#pragma unroll
            for (int p = 0; p < 8; ++p) {
                int kk = (p >> 2) * 16 + half * 8 + (p & 3) * 2;
                fb.d[p] = *(const unsigned int*)&ldsBT[cur][n * 32 + kk];
            }
#pragma unroll
            for (int ms = 0; ms < 2; ++ms)
                acc[ms][ns] = __builtin_amdgcn_wmma_f32_16x16x32_bf16(
                    false, fa[ms].b, false, fb.b, (short)0, acc[ms][ns],
                    false, false);
        }

        // ---- commit prefetched tiles to LDS -------------------------------
        if (doStage) {
#pragma unroll
            for (int e = 0; e < 8; ++e)
                *(unsigned int*)&ldsBT[nxt][(bc * 8 + e) * 32 + bk0] =
                    (unsigned)b0.s[e] | ((unsigned)b1.s[e] << 16);
#if !USE_TDM
#pragma unroll
            for (int it = 0; it < 2; ++it) {
                int chunk = tid + it * 256;
                int row = chunk >> 2, cc = chunk & 3;
                *(uint4*)&ldsA[nxt][row * 32 + cc * 8] = aR[it].v;
            }
#endif
        }
        __syncthreads();
    }

    // ---- epilogue ---------------------------------------------------------
#pragma unroll
    for (int ms = 0; ms < 2; ++ms) {
#pragma unroll
        for (int ns = 0; ns < 4; ++ns) {
#pragma unroll
            for (int i = 0; i < 8; ++i) {
                int r    = m0 + 32 * wm + 16 * ms + half * 8 + i;
                int ccol = n0 + 64 * wn + 16 * ns + l16;
                float v  = acc[ms][ns][i] + bias[ccol];
                if (mode == 0) {
                    ((float*)out)[(size_t)r * N + ccol] = v;
                } else if (mode == 1) {
                    int s = r / B_, b = r % B_;
                    int h = ccol / HD_, d = ccol % HD_;
                    ((unsigned short*)out)
                        [(((size_t)b * H_ + h) * S_ + s) * HD_ + d] =
                        f32_to_bf16(v * scale);
                } else {
                    ((unsigned short*)out)[(size_t)r * N + ccol] =
                        f32_to_bf16(v > 0.f ? v : 0.f);
                }
            }
        }
    }
}

// ---------------------------------------------------------------------------
// Flash attention (per (b,h) 64 q-rows per block, 128 threads = 4 waves).
// q/k/v are bf16 in (B,H,S,HD); q is pre-scaled by 1/sqrt(HD).
// K tile staged by TDM; V tile staged transposed by threads.
// out is fp32 in (S,B,E) layout.
// ---------------------------------------------------------------------------
__global__ __launch_bounds__(128)
void flash_attn(const unsigned short* __restrict__ q,
                const unsigned short* __restrict__ k,
                const unsigned short* __restrict__ v,
                const unsigned char* __restrict__ mask,
                float* __restrict__ out) {
    __shared__ __align__(16) unsigned short ldsK[32 * 64];    // [j][d]
    __shared__ __align__(16) unsigned short ldsVT[64 * 32];   // [d][j]
    __shared__ __align__(16) unsigned short ldsP[4][16 * 32]; // per wave [m][j]

    const int bh   = blockIdx.x >> 5;        // 0..31  (b*H + h)
    const int qblk = blockIdx.x & 31;
    const int b    = bh / H_;
    const int hh   = bh % H_;

    const int tid  = threadIdx.x;
    const int w    = tid >> 5;
    const int lane = tid & 31;
    const int half = lane >> 4;
    const int l16  = lane & 15;

    // V staging: 16 j-row pairs x 8 col-chunks
    const int vj0 = ((tid >> 3) & 15) * 2;
    const int vc  = tid & 7;

    const size_t head_base = (size_t)bh * S_ * HD_;

    // ---- load Q fragments for this wave's 16 rows (K=0..63 -> 2 frags)
    Frag16 qf[2];
    {
        int srow = qblk * QBLK + w * 16 + l16;
        const unsigned short* qr = q + head_base + (size_t)srow * HD_;
#pragma unroll
        for (int kf = 0; kf < 2; ++kf)
#pragma unroll
            for (int p = 0; p < 8; ++p) {
                int d = kf * 32 + (p >> 2) * 16 + half * 8 + (p & 3) * 2;
                qf[kf].d[p] = *(const unsigned int*)&qr[d];
            }
    }

    float mrow[8], lrow[8];
    v8f acc[4] = {{}, {}, {}, {}};
#pragma unroll
    for (int i = 0; i < 8; ++i) { mrow[i] = -1e30f; lrow[i] = 0.f; }

    for (int jt = 0; jt < S_ / 32; ++jt) {
        const int j0 = jt * 32;
        // ---- stage K (TDM) and V (transposed, paired b32 stores)
#if USE_TDM
        if (tid < 32)
            tdm_load_tile_2d(&ldsK[0], &k[head_base + (size_t)j0 * HD_],
                             64, 32, (unsigned)HD_, (unsigned)S_, (unsigned)HD_);
#else
#pragma unroll
        for (int it = 0; it < 2; ++it) {
            int chunk = tid + it * 128;
            int jr = chunk >> 3, cc = chunk & 7;
            U4 kv;
            kv.v = *(const uint4*)&k[head_base + (size_t)(j0 + jr) * HD_ + cc * 8];
            *(uint4*)&ldsK[jr * 64 + cc * 8] = kv.v;
        }
#endif
        {
            U4 v0r, v1r;
            v0r.v = *(const uint4*)&v[head_base + (size_t)(j0 + vj0 + 0) * HD_ + vc * 8];
            v1r.v = *(const uint4*)&v[head_base + (size_t)(j0 + vj0 + 1) * HD_ + vc * 8];
#pragma unroll
            for (int e = 0; e < 8; ++e)
                *(unsigned int*)&ldsVT[(vc * 8 + e) * 32 + vj0] =
                    (unsigned)v0r.s[e] | ((unsigned)v1r.s[e] << 16);
        }
#if USE_TDM
        if (tid < 32) __builtin_amdgcn_s_wait_tensorcnt(0);
#endif
        __syncthreads();

        // ---- scores: two 16x16 fragments over K=64 (2 chained WMMAs each)
        v8f s0 = {}, s1 = {};
#pragma unroll
        for (int ns = 0; ns < 2; ++ns) {
#pragma unroll
            for (int kf = 0; kf < 2; ++kf) {
                Frag16 fb;
                int j = ns * 16 + l16;
#pragma unroll
                for (int p = 0; p < 8; ++p) {
                    int d = kf * 32 + (p >> 2) * 16 + half * 8 + (p & 3) * 2;
                    fb.d[p] = *(const unsigned int*)&ldsK[j * 64 + d];
                }
                if (ns == 0)
                    s0 = __builtin_amdgcn_wmma_f32_16x16x32_bf16(
                        false, qf[kf].b, false, fb.b, (short)0, s0, false, false);
                else
                    s1 = __builtin_amdgcn_wmma_f32_16x16x32_bf16(
                        false, qf[kf].b, false, fb.b, (short)0, s1, false, false);
            }
        }

        // ---- key padding mask (one column per lane per subtile)
        {
            unsigned char m0b = mask[(size_t)b * S_ + j0 + l16];
            unsigned char m1b = mask[(size_t)b * S_ + j0 + 16 + l16];
            if (m0b) {
#pragma unroll
                for (int i = 0; i < 8; ++i) s0[i] = -1e30f;
            }
            if (m1b) {
#pragma unroll
                for (int i = 0; i < 8; ++i) s1[i] = -1e30f;
            }
        }

        // ---- online softmax update (row = half*8 + i, 16 lanes per row)
#pragma unroll
        for (int i = 0; i < 8; ++i) {
            float rm = fmaxf(s0[i], s1[i]);
#pragma unroll
            for (int off = 1; off < 16; off <<= 1)
                rm = fmaxf(rm, __shfl_xor(rm, off, 32));
            float mn = fmaxf(mrow[i], rm);
            float al = __expf(mrow[i] - mn);
            float p0 = __expf(s0[i] - mn);
            float p1 = __expf(s1[i] - mn);
            float sm = p0 + p1;
#pragma unroll
            for (int off = 1; off < 16; off <<= 1)
                sm += __shfl_xor(sm, off, 32);
            lrow[i] = lrow[i] * al + sm;
            mrow[i] = mn;
#pragma unroll
            for (int f = 0; f < 4; ++f) acc[f][i] *= al;
            int row = half * 8 + i;
            ldsP[w][row * 32 + l16]      = f32_to_bf16(p0);
            ldsP[w][row * 32 + 16 + l16] = f32_to_bf16(p1);
        }

        // ---- P fragment (16x32) and acc += P * V
        Frag16 fp;
#pragma unroll
        for (int p = 0; p < 8; ++p) {
            int kk = (p >> 2) * 16 + half * 8 + (p & 3) * 2;
            fp.d[p] = *(const unsigned int*)&ldsP[w][l16 * 32 + kk];
        }
#pragma unroll
        for (int f = 0; f < 4; ++f) {
            Frag16 fv;
            int dcol = f * 16 + l16;
#pragma unroll
            for (int p = 0; p < 8; ++p) {
                int j = (p >> 2) * 16 + half * 8 + (p & 3) * 2;
                fv.d[p] = *(const unsigned int*)&ldsVT[dcol * 32 + j];
            }
            acc[f] = __builtin_amdgcn_wmma_f32_16x16x32_bf16(
                false, fp.b, false, fv.b, (short)0, acc[f], false, false);
        }
        __syncthreads();
    }

    // ---- finalize: out[s,b, hh*HD + d] = acc / l
#pragma unroll
    for (int i = 0; i < 8; ++i) {
        float rl = lrow[i] > 0.f ? 1.f / lrow[i] : 0.f;
        int sg = qblk * QBLK + w * 16 + half * 8 + i;
#pragma unroll
        for (int f = 0; f < 4; ++f) {
            int e = hh * HD_ + f * 16 + l16;
            out[((size_t)sg * B_ + b) * E_ + e] = acc[f][i] * rl;
        }
    }
}

// ---------------------------------------------------------------------------
// y = LayerNorm(res + delta) * g + b ; writes fp32 and optional bf16 copy.
// ---------------------------------------------------------------------------
__global__ __launch_bounds__(256)
void add_ln(const float* __restrict__ res, const float* __restrict__ delta,
            const float* __restrict__ g, const float* __restrict__ bb,
            float* __restrict__ out, unsigned short* __restrict__ outb) {
    __shared__ float red[256];
    const size_t t = blockIdx.x;
    const int tid = threadIdx.x;

    float vals[4];
    float s = 0.f;
#pragma unroll
    for (int kk = 0; kk < 4; ++kk) {
        int e = tid + kk * 256;
        float vv = res[t * E_ + e] + delta[t * E_ + e];
        vals[kk] = vv;
        s += vv;
    }
    red[tid] = s;
    __syncthreads();
    for (int off = 128; off > 0; off >>= 1) {
        if (tid < off) red[tid] += red[tid + off];
        __syncthreads();
    }
    float mean = red[0] * (1.f / E_);
    __syncthreads();

    float sq = 0.f;
#pragma unroll
    for (int kk = 0; kk < 4; ++kk) {
        float d = vals[kk] - mean;
        sq += d * d;
    }
    red[tid] = sq;
    __syncthreads();
    for (int off = 128; off > 0; off >>= 1) {
        if (tid < off) red[tid] += red[tid + off];
        __syncthreads();
    }
    float rstd = rsqrtf(red[0] * (1.f / E_) + 1e-5f);

#pragma unroll
    for (int kk = 0; kk < 4; ++kk) {
        int e = tid + kk * 256;
        float y = (vals[kk] - mean) * rstd * g[e] + bb[e];
        out[t * E_ + e] = y;
        if (outb) outb[t * E_ + e] = f32_to_bf16(y);
    }
}

// ---------------------------------------------------------------------------
// Host orchestration
// ---------------------------------------------------------------------------
extern "C" void kernel_launch(void* const* d_in, const int* in_sizes, int n_in,
                              void* d_out, int out_size, void* d_ws, size_t ws_size,
                              hipStream_t stream) {
    const float* x    = (const float*)d_in[0];
    const unsigned char* mask = (const unsigned char*)d_in[1];
    const float* Wq = (const float*)d_in[2];  const float* bq = (const float*)d_in[3];
    const float* Wk = (const float*)d_in[4];  const float* bk = (const float*)d_in[5];
    const float* Wv = (const float*)d_in[6];  const float* bv = (const float*)d_in[7];
    const float* Wo = (const float*)d_in[8];  const float* bo = (const float*)d_in[9];
    const float* ln1_g = (const float*)d_in[10]; const float* ln1_b = (const float*)d_in[11];
    const float* W1 = (const float*)d_in[12]; const float* b1 = (const float*)d_in[13];
    const float* W2 = (const float*)d_in[14]; const float* b2 = (const float*)d_in[15];
    const float* ln2_g = (const float*)d_in[16]; const float* ln2_b = (const float*)d_in[17];
    float* outp = (float*)d_out;

    char* ws = (char*)d_ws;
    size_t off = 0;
    auto alloc = [&](size_t bytes) -> char* {
        char* p = ws + off;
        off += (bytes + 255) & ~(size_t)255;
        return p;
    };
    unsigned short* xb   = (unsigned short*)alloc((size_t)NTOK * E_ * 2);
    unsigned short* wqb  = (unsigned short*)alloc((size_t)E_ * E_ * 2);
    unsigned short* wkb  = (unsigned short*)alloc((size_t)E_ * E_ * 2);
    unsigned short* wvb  = (unsigned short*)alloc((size_t)E_ * E_ * 2);
    unsigned short* wob  = (unsigned short*)alloc((size_t)E_ * E_ * 2);
    unsigned short* w1b  = (unsigned short*)alloc((size_t)E_ * FFN_ * 2);
    unsigned short* w2b  = (unsigned short*)alloc((size_t)FFN_ * E_ * 2);
    unsigned short* qb   = (unsigned short*)alloc((size_t)NTOK * E_ * 2);
    unsigned short* kbuf = (unsigned short*)alloc((size_t)NTOK * E_ * 2);
    unsigned short* vb   = (unsigned short*)alloc((size_t)NTOK * E_ * 2);
    float* attn          = (float*)alloc((size_t)NTOK * E_ * 4);
    unsigned short* attnb= (unsigned short*)alloc((size_t)NTOK * E_ * 2);
    float* hbuf          = (float*)alloc((size_t)NTOK * E_ * 4);
    unsigned short* hb   = (unsigned short*)alloc((size_t)NTOK * E_ * 2);
    unsigned short* f1b  = (unsigned short*)alloc((size_t)NTOK * FFN_ * 2);
    float* f2            = (float*)alloc((size_t)NTOK * E_ * 4);

    // ---- converts
    cvt_f32_bf16<<<2048, 256, 0, stream>>>(x,  xb,  NTOK * E_);
    cvt_f32_bf16<<<1024, 256, 0, stream>>>(Wq, wqb, E_ * E_);
    cvt_f32_bf16<<<1024, 256, 0, stream>>>(Wk, wkb, E_ * E_);
    cvt_f32_bf16<<<1024, 256, 0, stream>>>(Wv, wvb, E_ * E_);
    cvt_f32_bf16<<<1024, 256, 0, stream>>>(Wo, wob, E_ * E_);
    cvt_f32_bf16<<<2048, 256, 0, stream>>>(W1, w1b, E_ * FFN_);
    cvt_f32_bf16<<<2048, 256, 0, stream>>>(W2, w2b, FFN_ * E_);

    // ---- QKV projections (bf16 out, scattered to (B,H,S,HD); q pre-scaled)
    dim3 gE(E_ / 128, NTOK / 128);
    gemm_bf16<<<gE, 256, 0, stream>>>(xb, wqb, bq, NTOK, E_, E_, 1, 0.125f, qb);
    gemm_bf16<<<gE, 256, 0, stream>>>(xb, wkb, bk, NTOK, E_, E_, 1, 1.0f, kbuf);
    gemm_bf16<<<gE, 256, 0, stream>>>(xb, wvb, bv, NTOK, E_, E_, 1, 1.0f, vb);

    // ---- fused flash attention
    flash_attn<<<B_ * H_ * NQB, 128, 0, stream>>>(qb, kbuf, vb, mask, attn);

    // ---- output projection + residual + LN1
    cvt_f32_bf16<<<2048, 256, 0, stream>>>(attn, attnb, NTOK * E_);
    gemm_bf16<<<gE, 256, 0, stream>>>(attnb, wob, bo, NTOK, E_, E_, 0, 1.0f, f2);
    add_ln<<<NTOK, 256, 0, stream>>>(x, f2, ln1_g, ln1_b, hbuf, hb);

    // ---- FFN
    dim3 gF1(FFN_ / 128, NTOK / 128);
    gemm_bf16<<<gF1, 256, 0, stream>>>(hb, w1b, b1, NTOK, FFN_, E_, 2, 1.0f, f1b);
    gemm_bf16<<<gE, 256, 0, stream>>>(f1b, w2b, b2, NTOK, E_, FFN_, 0, 1.0f, f2);
    add_ln<<<NTOK, 256, 0, stream>>>(hbuf, f2, ln2_g, ln2_b, outp, nullptr);
}
